// LocalDotAttention_13709535609372
// MI455X (gfx1250) — matrix operand order repaired and checked by reading, not compiled, not run on hardware
//
#include <hip/hip_runtime.h>
#include <stdint.h>

// ---- problem constants (match reference) ----
#define B_  4
#define H_  16
#define S_  2048
#define D_  64      // DK == DV == 64
#define BQ  128     // q rows per block (8 waves x 16 rows)
#define BK  32      // keys per inner iteration

#define LOG2E 1.4426950408889634f

typedef _Float16 v16h __attribute__((ext_vector_type(16)));
typedef _Float16 v8h  __attribute__((ext_vector_type(8)));
typedef float    v8f  __attribute__((ext_vector_type(8)));

// Build a 16-half fragment from two contiguous 8-half (16B) LDS chunks.
__device__ __forceinline__ v16h join8(const _Float16* p0, const _Float16* p1) {
    v8h a = *(const v8h*)p0;
    v8h b = *(const v8h*)p1;
    return __builtin_shufflevector(a, b, 0,1,2,3,4,5,6,7,8,9,10,11,12,13,14,15);
}

// Raw IEEE maximumNumber without clang's canonicalize pre-ops (data is NaN-free).
__device__ __forceinline__ float fmax_raw(float a, float b) {
    float d;
    asm("v_max_num_f32 %0, %1, %2" : "=v"(d) : "v"(a), "v"(b));
    return d;
}

// Fused VALU butterfly steps: DPP16 ROW_XMASK applied directly as the src0
// modifier of v_max_num/v_add -> 1 instruction per step (VALU deps are
// hardware-interlocked on CDNA5; s_delay_alu is perf-only per ISA ch.8).
__device__ __forceinline__ float rowmax16(float x) {
    asm("v_max_num_f32_dpp %0, %0, %0 row_xmask:1 row_mask:0xf bank_mask:0xf bound_ctrl:1" : "+v"(x));
    asm("v_max_num_f32_dpp %0, %0, %0 row_xmask:2 row_mask:0xf bank_mask:0xf bound_ctrl:1" : "+v"(x));
    asm("v_max_num_f32_dpp %0, %0, %0 row_xmask:4 row_mask:0xf bank_mask:0xf bound_ctrl:1" : "+v"(x));
    asm("v_max_num_f32_dpp %0, %0, %0 row_xmask:8 row_mask:0xf bank_mask:0xf bound_ctrl:1" : "+v"(x));
    return x;
}
__device__ __forceinline__ float rowsum16(float x) {
    asm("v_add_f32_dpp %0, %0, %0 row_xmask:1 row_mask:0xf bank_mask:0xf bound_ctrl:1" : "+v"(x));
    asm("v_add_f32_dpp %0, %0, %0 row_xmask:2 row_mask:0xf bank_mask:0xf bound_ctrl:1" : "+v"(x));
    asm("v_add_f32_dpp %0, %0, %0 row_xmask:4 row_mask:0xf bank_mask:0xf bound_ctrl:1" : "+v"(x));
    asm("v_add_f32_dpp %0, %0, %0 row_xmask:8 row_mask:0xf bank_mask:0xf bound_ctrl:1" : "+v"(x));
    return x;
}

// Raw v_exp_f32 (base-2 exponent, no log2e pre-multiply).
__device__ __forceinline__ float exp2f_fast(float x) {
    return __builtin_amdgcn_exp2f(x);
}

__global__ __launch_bounds__(256)
void attn_fwd_kernel(const float* __restrict__ Qg,
                     const float* __restrict__ Kg,
                     const float* __restrict__ Vg,
                     const uint8_t* __restrict__ Mg,
                     float* __restrict__ Og)
{
    __shared__ __align__(16) _Float16 sQ [BQ * D_];       // 16 KB  q tile [row][d], pre-scaled by log2e/8
    __shared__ __align__(16) _Float16 sK [2][BK * D_];    //  8 KB  k tiles (ping-pong) [key][d]
    __shared__ __align__(16) _Float16 sVt[2][D_ * BK];    //  8 KB  v tiles (ping-pong) TRANSPOSED [d][key]
    __shared__ __align__(16) _Float16 sP [8 * 16 * BK];   //  8 KB  per-wave P relayout scratch
    __shared__ float sMb[S_];                             //  8 KB  mask bias: 0 or -1e9 (base-2 logit units)

    const int tid  = threadIdx.x;
    const int lane = tid & 31;
    const int wv   = tid >> 5;        // wave id 0..7
    const int r    = lane & 15;       // lane-within-half
    const int hsel = lane >> 4;       // half-wave select

    const int qt = blockIdx.x;        // q tile of 128 rows
    const int hh = blockIdx.y;
    const int bb = blockIdx.z;

    const size_t head = (size_t)bb * H_ + hh;
    const float*   Qh = Qg + (head * S_ + (size_t)qt * BQ) * D_;
    const float*   Kh = Kg + head * S_ * D_;
    const float*   Vh = Vg + head * S_ * D_;
    const uint8_t* Mh = Mg + (size_t)bb * S_;
    float*         Oh = Og + (head * S_ + (size_t)qt * BQ) * D_;

    // Gaussian decay in base-2 domain: exp(-d^2 * 2/S^2) = exp2(-d^2 * 2*log2e/S^2)
    const float C2 = (2.0f * LOG2E) / ((float)S_ * (float)S_);

    // ---- Q tile: 128x64 fp32 -> f16 LDS, folding (1/sqrt(64))*log2(e) into Q ----
    {
        const float qs = 0.125f * LOG2E;
        const int row0 = tid >> 1;
        const int col0 = (tid & 1) * 32;
        const float4* src = (const float4*)(Qh + (size_t)row0 * D_ + col0);
        _Float16* dst = &sQ[row0 * D_ + col0];
        #pragma unroll
        for (int i = 0; i < 8; ++i) {
            float4 q = src[i];
            dst[i*4+0] = (_Float16)(q.x * qs); dst[i*4+1] = (_Float16)(q.y * qs);
            dst[i*4+2] = (_Float16)(q.z * qs); dst[i*4+3] = (_Float16)(q.w * qs);
        }
    }
    // ---- expand key mask into float bias table once ----
    #pragma unroll
    for (int i = 0; i < S_ / 256; ++i) {
        const int idx = tid + i * 256;
        sMb[idx] = Mh[idx] ? -1e9f : 0.0f;
    }

    // K/V tile loader mapping: 2048 fp32 / 256 threads = 8 per thread
    const int lrow = tid >> 3;          // 0..31 (key within tile)
    const int lcol = (tid & 7) * 8;     // 0..56 (d)

    // ---- prologue: fetch tile 0 into buffer 0 ----
    {
        const float4* ks = (const float4*)(Kh + (size_t)lrow * D_ + lcol);
        const float4* vs = (const float4*)(Vh + (size_t)lrow * D_ + lcol);
        float4 k0 = ks[0], k1 = ks[1];
        float4 v0 = vs[0], v1 = vs[1];
        _Float16* kd = &sK[0][lrow * D_ + lcol];
        kd[0] = (_Float16)k0.x; kd[1] = (_Float16)k0.y;
        kd[2] = (_Float16)k0.z; kd[3] = (_Float16)k0.w;
        kd[4] = (_Float16)k1.x; kd[5] = (_Float16)k1.y;
        kd[6] = (_Float16)k1.z; kd[7] = (_Float16)k1.w;
        float vv[8] = { v0.x, v0.y, v0.z, v0.w, v1.x, v1.y, v1.z, v1.w };
        #pragma unroll
        for (int i = 0; i < 8; ++i)
            sVt[0][(lcol + i) * BK + lrow] = (_Float16)vv[i];
    }
    __syncthreads();

    // ---- Q A-fragments for this wave's 16 rows (resident in VGPRs) ----
    const _Float16* qrow = &sQ[(wv * 16 + r) * D_];
    const v16h qa0 = join8(qrow +      16 * hsel, qrow +      16 * hsel + 8);  // d 0..31
    const v16h qa1 = join8(qrow + 32 + 16 * hsel, qrow + 32 + 16 * hsel + 8);  // d 32..63

    v8f acc0 = {}, acc1 = {}, acc2 = {}, acc3 = {};
    float m_i[8], l_i[8];
    #pragma unroll
    for (int v = 0; v < 8; ++v) { m_i[v] = -1e30f; l_i[v] = 0.0f; }

    const int qrow_i = qt * BQ + wv * 16 + 8 * hsel;    // + v per element
    _Float16* sPw = &sP[wv * 16 * BK];

    int cur = 0;
    for (int kb = 0; kb < S_; kb += BK) {
        const bool hv = (kb + BK) < S_;

        // ---- issue next tile's global loads before the barrier (overlap with compute) ----
        float4 nk0, nk1, nv0, nv1;
        if (hv) {
            const float4* ks = (const float4*)(Kh + (size_t)(kb + BK + lrow) * D_ + lcol);
            const float4* vs = (const float4*)(Vh + (size_t)(kb + BK + lrow) * D_ + lcol);
            nk0 = ks[0]; nk1 = ks[1];
            nv0 = vs[0]; nv1 = vs[1];
            if (kb + 2 * BK < S_) {  // global_prefetch_b8 two tiles ahead
                __builtin_prefetch(Kh + (size_t)(kb + 2 * BK + lrow) * D_ + lcol, 0, 0);
                __builtin_prefetch(Vh + (size_t)(kb + 2 * BK + lrow) * D_ + lcol, 0, 0);
            }
        }

        __syncthreads();   // buf[cur] ready; everyone done reading buf[cur^1]

        // ---- scores (base-2 logits): S(16q x 32k) = Qs @ K^T via 4 WMMA ----
        const _Float16* k0p = &sK[cur][r * D_];
        const _Float16* k1p = &sK[cur][(16 + r) * D_];
        v16h kb00 = join8(k0p +      16 * hsel, k0p +      16 * hsel + 8);
        v16h kb01 = join8(k0p + 32 + 16 * hsel, k0p + 32 + 16 * hsel + 8);
        v16h kb10 = join8(k1p +      16 * hsel, k1p +      16 * hsel + 8);
        v16h kb11 = join8(k1p + 32 + 16 * hsel, k1p + 32 + 16 * hsel + 8);

        v8f cz = {};
        v8f s0 = __builtin_amdgcn_wmma_f32_16x16x32_f16(false, qa0, false, kb00, (short)0, cz, false, false);
        s0     = __builtin_amdgcn_wmma_f32_16x16x32_f16(false, qa1, false, kb01, (short)0, s0, false, false);
        v8f s1 = __builtin_amdgcn_wmma_f32_16x16x32_f16(false, qa0, false, kb10, (short)0, cz, false, false);
        s1     = __builtin_amdgcn_wmma_f32_16x16x32_f16(false, qa1, false, kb11, (short)0, s1, false, false);

        // ---- Gaussian decay factors: independent of WMMA results -> fills the
        //      f16 WMMA->VALU hazard slots instead of v_nop padding ----
        const float fb0 = (float)(qrow_i - (kb + r));        // (i - j) base, + v per row
        const float fb1 = fb0 - 16.0f;
        float dec0[8], dec1[8];
        #pragma unroll
        for (int v = 0; v < 8; ++v) {
            float t0 = fb0 + (float)v;
            float t1 = fb1 + (float)v;
            dec0[v] = exp2f_fast(-(t0 * t0) * C2);
            dec1[v] = exp2f_fast(-(t1 * t1) * C2);
        }

        // ---- mask bias (LDS table) + online softmax (fused DPP reductions) ----
        const float bias0 = sMb[kb + r];
        const float bias1 = sMb[kb + 16 + r];

        float p0[8], p1[8], sf[8];
        #pragma unroll
        for (int v = 0; v < 8; ++v) {
            float a0 = s0[v] + bias0;
            float a1 = s1[v] + bias1;
            float mx = rowmax16(fmax_raw(a0, a1));
            float mn = fmax_raw(m_i[v], mx);
            float f  = exp2f_fast(m_i[v] - mn);
            float e0 = exp2f_fast(a0 - mn);
            float e1 = exp2f_fast(a1 - mn);
            float ls = rowsum16(e0 + e1);      // normalizer excludes decay (per reference)
            l_i[v] = l_i[v] * f + ls;
            m_i[v] = mn;
            sf[v]  = f;
            p0[v] = e0 * dec0[v];
            p1[v] = e1 * dec1[v];
        }
        #pragma unroll
        for (int v = 0; v < 8; ++v) {
            acc0[v] *= sf[v]; acc1[v] *= sf[v];
            acc2[v] *= sf[v]; acc3[v] *= sf[v];
        }

        // ---- relayout P (C layout -> A layout) through per-wave LDS patch ----
        #pragma unroll
        for (int v = 0; v < 8; ++v) {
            const int prow = v + 8 * hsel;
            sPw[prow * BK + r]      = (_Float16)p0[v];
            sPw[prow * BK + 16 + r] = (_Float16)p1[v];
        }
        asm volatile("s_wait_dscnt 0" ::: "memory");  // DS in-order per wave

        const _Float16* pr = &sPw[r * BK];
        v16h pa = join8(pr + 16 * hsel, pr + 16 * hsel + 8);

        // ---- O += P(16x32) @ V(32x64): 4 WMMA over DV tiles ----
        {
            const _Float16* vp0 = &sVt[cur][( 0 + r) * BK + 16 * hsel];
            const _Float16* vp1 = &sVt[cur][(16 + r) * BK + 16 * hsel];
            const _Float16* vp2 = &sVt[cur][(32 + r) * BK + 16 * hsel];
            const _Float16* vp3 = &sVt[cur][(48 + r) * BK + 16 * hsel];
            v16h vb0 = join8(vp0, vp0 + 8);
            v16h vb1 = join8(vp1, vp1 + 8);
            v16h vb2 = join8(vp2, vp2 + 8);
            v16h vb3 = join8(vp3, vp3 + 8);
            acc0 = __builtin_amdgcn_wmma_f32_16x16x32_f16(false, pa, false, vb0, (short)0, acc0, false, false);
            acc1 = __builtin_amdgcn_wmma_f32_16x16x32_f16(false, pa, false, vb1, (short)0, acc1, false, false);
            acc2 = __builtin_amdgcn_wmma_f32_16x16x32_f16(false, pa, false, vb2, (short)0, acc2, false, false);
            acc3 = __builtin_amdgcn_wmma_f32_16x16x32_f16(false, pa, false, vb3, (short)0, acc3, false, false);
        }

        // ---- stage next tile into the other LDS buffer (loads already in flight) ----
        if (hv) {
            const int nxt = cur ^ 1;
            _Float16* kd = &sK[nxt][lrow * D_ + lcol];
            kd[0] = (_Float16)nk0.x; kd[1] = (_Float16)nk0.y;
            kd[2] = (_Float16)nk0.z; kd[3] = (_Float16)nk0.w;
            kd[4] = (_Float16)nk1.x; kd[5] = (_Float16)nk1.y;
            kd[6] = (_Float16)nk1.z; kd[7] = (_Float16)nk1.w;
            float vv[8] = { nv0.x, nv0.y, nv0.z, nv0.w, nv1.x, nv1.y, nv1.z, nv1.w };
            #pragma unroll
            for (int i = 0; i < 8; ++i)
                sVt[nxt][(lcol + i) * BK + lrow] = (_Float16)vv[i];
        }
        cur ^= 1;
    }

    // ---- epilogue: O = acc / l ----
    #pragma unroll
    for (int v = 0; v < 8; ++v) {
        const float inv = 1.0f / l_i[v];
        const int row = wv * 16 + v + 8 * hsel;
        float* op = Oh + (size_t)row * D_ + r;
        op[ 0] = acc0[v] * inv;
        op[16] = acc1[v] * inv;
        op[32] = acc2[v] * inv;
        op[48] = acc3[v] * inv;
    }
}

extern "C" void kernel_launch(void* const* d_in, const int* in_sizes, int n_in,
                              void* d_out, int out_size, void* d_ws, size_t ws_size,
                              hipStream_t stream) {
    (void)in_sizes; (void)n_in; (void)out_size; (void)d_ws; (void)ws_size;
    const float*   Q = (const float*)d_in[0];
    const float*   K = (const float*)d_in[1];
    const float*   V = (const float*)d_in[2];
    const uint8_t* M = (const uint8_t*)d_in[3];   // jnp.bool_ -> 1 byte per element
    float* O = (float*)d_out;

    dim3 grid(S_ / BQ, H_, B_);   // (16, 16, 4)
    dim3 block(256);              // 8 wave32 waves
    attn_fwd_kernel<<<grid, block, 0, stream>>>(Q, K, V, M, O);
}
